// SubjectConditionalLinear_77360950935845
// MI455X (gfx1250) — matrix operand order repaired and checked by reading, not compiled, not run on hardware
//
#include <hip/hip_runtime.h>
#include <stdint.h>

// Problem constants (match the reference)
#define S_ 8
#define O_ 1024
#define H_ 1024
#define B_ 32
#define N_ 512

// Tiling
#define BM 128          // rows (n) per block
#define BN 128          // cols (o) per block
#define BK 32           // K per stage (= WMMA bf16 K)
#define LDSS 36         // LDS row stride in f32 (32 + 4 pad, 144B, 16B aligned)
#define KSTEPS (H_ / BK)

typedef __attribute__((ext_vector_type(16))) __bf16 v16bf;
typedef __attribute__((ext_vector_type(8)))  float  v8f;
typedef __attribute__((ext_vector_type(4)))  float  f4;
typedef __attribute__((ext_vector_type(4)))  int    v4i;

#define AS1 __attribute__((address_space(1)))
#define AS3 __attribute__((address_space(3)))

#if __has_builtin(__builtin_amdgcn_global_load_async_to_lds_b128)
#define ASYNC_LDS 1
#else
#define ASYNC_LDS 0
#endif

__device__ __forceinline__ void cp_async16(const float* g, float* l) {
#if ASYNC_LDS
  __builtin_amdgcn_global_load_async_to_lds_b128((AS1 v4i*)g, (AS3 v4i*)l, 0, 0);
#else
  *(f4*)l = *(const f4*)g;
#endif
}

__device__ __forceinline__ void async_wait8() {
#if ASYNC_LDS
 #if __has_builtin(__builtin_amdgcn_s_wait_asynccnt)
  __builtin_amdgcn_s_wait_asynccnt(8);
 #else
  asm volatile("s_wait_asynccnt 8" ::: "memory");
 #endif
#endif
}

__device__ __forceinline__ void async_wait0() {
#if ASYNC_LDS
 #if __has_builtin(__builtin_amdgcn_s_wait_asynccnt)
  __builtin_amdgcn_s_wait_asynccnt(0);
 #else
  asm volatile("s_wait_asynccnt 0" ::: "memory");
 #endif
#endif
}

__device__ __forceinline__ __bf16 cvtbf(float f) { return (__bf16)f; }

__global__ __launch_bounds__(256)
void scl_wmma_kernel(const float* __restrict__ x, const int* __restrict__ sid,
                     const float* __restrict__ w, const float* __restrict__ bias,
                     float* __restrict__ y)
{
  __shared__ float sA[2][BM * LDSS];   // x tile   (f32, 18KB each)
  __shared__ float sB[2][BN * LDSS];   // W tile   (f32, 18KB each)

  const int tid  = threadIdx.x;
  const int lane = tid & 31;
  const int wave = tid >> 5;
  const int half = lane >> 4;          // 0 | 1  (wave32 half)
  const int lrow = lane & 15;
  const int og   = wave & 3;           // o-group: 32 cols each
  const int mg   = wave >> 2;          // m-group: 64 rows each

  const int oblk = blockIdx.x * BN;
  const int nblk = blockIdx.y * BM;
  const int b    = blockIdx.z;
  const int s    = sid[b];

  const float* xg = x + ((size_t)b * N_ + nblk) * H_;
  const float* wg = w + ((size_t)s * O_ + oblk) * H_;

  v8f acc[4][2];
  const v8f vzero = {0.f,0.f,0.f,0.f,0.f,0.f,0.f,0.f};
#pragma unroll
  for (int i = 0; i < 4; ++i)
#pragma unroll
    for (int j = 0; j < 2; ++j) acc[i][j] = vzero;

  // Stage loader: 128 rows x 32 f32 per tile = 1024 16B chunks; 256 thr x 4.
  auto issue = [&](int ks) {
    const float* xa = xg + ks * BK;
    const float* wa = wg + ks * BK;
    float* la = sA[ks & 1];
    float* lb = sB[ks & 1];
#pragma unroll
    for (int i = 0; i < 4; ++i) {
      int c   = tid + 256 * i;
      int row = c >> 3;
      int cc  = (c & 7) * 4;           // f32 column within the 32-wide tile
      cp_async16(xa + (size_t)row * H_ + cc, la + row * LDSS + cc);
      cp_async16(wa + (size_t)row * H_ + cc, lb + row * LDSS + cc);
    }
  };

  issue(0);

  for (int ks = 0; ks < KSTEPS; ++ks) {
    if (ks + 1 < KSTEPS) { issue(ks + 1); async_wait8(); }
    else                 { async_wait0(); }
    __syncthreads();

    const float* la = sA[ks & 1];
    const float* lb = sB[ks & 1];

    // A fragments (16x32 bf16): lanes 0-15 -> K 0-7 & 16-23; lanes 16-31 -> K 8-15 & 24-31
    v16bf af[4];
#pragma unroll
    for (int mt = 0; mt < 4; ++mt) {
      const float* p = la + (mg * 64 + mt * 16 + lrow) * LDSS + half * 8;
      f4 a0 = *(const f4*)(p);
      f4 a1 = *(const f4*)(p + 4);
      f4 a2 = *(const f4*)(p + 16);
      f4 a3 = *(const f4*)(p + 20);
      v16bf v;
#pragma unroll
      for (int e = 0; e < 4; ++e) {
        v[e]      = cvtbf(a0[e]);
        v[4 + e]  = cvtbf(a1[e]);
        v[8 + e]  = cvtbf(a2[e]);
        v[12 + e] = cvtbf(a3[e]);
      }
      af[mt] = v;
    }

    // B fragments (32x16 bf16): lane holds column N=lrow (= one W row),
    // lanes 0-15 -> K 0-7 & 16-23, lanes 16-31 -> K 8-15 & 24-31.
    v16bf bf[2];
#pragma unroll
    for (int ot = 0; ot < 2; ++ot) {
      const float* p = lb + (og * 32 + ot * 16 + lrow) * LDSS + half * 8;
      f4 b0 = *(const f4*)(p);
      f4 b1 = *(const f4*)(p + 4);
      f4 b2 = *(const f4*)(p + 16);
      f4 b3 = *(const f4*)(p + 20);
      v16bf v;
#pragma unroll
      for (int e = 0; e < 4; ++e) {
        v[e]      = cvtbf(b0[e]);
        v[4 + e]  = cvtbf(b1[e]);
        v[8 + e]  = cvtbf(b2[e]);
        v[12 + e] = cvtbf(b3[e]);
      }
      bf[ot] = v;
    }

#pragma unroll
    for (int mt = 0; mt < 4; ++mt)
#pragma unroll
      for (int ot = 0; ot < 2; ++ot)
        acc[mt][ot] = __builtin_amdgcn_wmma_f32_16x16x32_bf16(
            false, af[mt], false, bf[ot], (short)0, acc[mt][ot], false, false);

    __syncthreads();
  }

  // Epilogue: gathered bias add + coalesced f32 stores.
  // C/D layout: lane covers column N=lrow, rows M = half*8 + r (VGPR r).
#pragma unroll
  for (int ot = 0; ot < 2; ++ot) {
    int o = oblk + og * 32 + ot * 16 + lrow;
    float bv = bias[s * O_ + o];
#pragma unroll
    for (int mt = 0; mt < 4; ++mt) {
      int m = nblk + mg * 64 + mt * 16 + half * 8;
      float* yp = y + ((size_t)b * N_ + m) * O_ + o;
#pragma unroll
      for (int r = 0; r < 8; ++r)
        yp[(size_t)r * O_] = acc[mt][ot][r] + bv;
    }
  }
}

extern "C" void kernel_launch(void* const* d_in, const int* in_sizes, int n_in,
                              void* d_out, int out_size, void* d_ws, size_t ws_size,
                              hipStream_t stream) {
  (void)in_sizes; (void)n_in; (void)out_size; (void)d_ws; (void)ws_size;
  const float* x    = (const float*)d_in[0];
  const int*   sid  = (const int*)d_in[1];
  const float* w    = (const float*)d_in[2];
  const float* bias = (const float*)d_in[3];
  float*       y    = (float*)d_out;

  dim3 grid(O_ / BN, N_ / BM, B_);
  dim3 block(256);
  scl_wmma_kernel<<<grid, block, 0, stream>>>(x, sid, w, bias, y);
}
